// GLORN_23965917512363
// MI455X (gfx1250) — matrix operand order repaired and checked by reading, not compiled, not run on hardware
//
#include <hip/hip_runtime.h>

// ---------------------------------------------------------------------------
// Types / helpers
// ---------------------------------------------------------------------------
typedef __attribute__((ext_vector_type(16))) __bf16          v16bf;
typedef __attribute__((ext_vector_type(8)))  float           v8f;
typedef __attribute__((ext_vector_type(8)))  unsigned short  ushort8;
typedef __attribute__((ext_vector_type(4)))  unsigned short  u16x4;

union BF16x16 { unsigned short u[16]; v16bf v; };

__device__ __forceinline__ unsigned short f2bf(float f) {
    unsigned int u = __float_as_uint(f);
    u += 0x7FFFu + ((u >> 16) & 1u);          // round-to-nearest-even
    return (unsigned short)(u >> 16);
}
__device__ __forceinline__ float bf2f(unsigned short h) {
    return __uint_as_float(((unsigned int)h) << 16);
}
__device__ __forceinline__ float lrelu(float v) { return v > 0.f ? v : 0.1f * v; }

// ---------------------------------------------------------------------------
// fp32 -> bf16 conversion (vec4)
// ---------------------------------------------------------------------------
__global__ void k_cvt4(const float* __restrict__ x, unsigned short* __restrict__ y, int n4) {
    int i = blockIdx.x * blockDim.x + threadIdx.x;
    if (i < n4) {
        float4 v = *(const float4*)(x + (size_t)i * 4);
        u16x4 o;
        o[0] = f2bf(v.x); o[1] = f2bf(v.y); o[2] = f2bf(v.z); o[3] = f2bf(v.w);
        *(u16x4*)(y + (size_t)i * 4) = o;
    }
}

// ---------------------------------------------------------------------------
// bf16 WMMA GEMM:  C[M,N] = act( rowscale[M] * (A[M,K] @ B[K,N]) + bias[N] )
// Block tile 128x64, K-step 32, 8 waves; wave (wr,wc) owns rows wr*32..+32,
// cols wc*32..+32 -> four 16x16 WMMA accumulators (4 WMMA / wave / K-step).
// Tile fills are per-chunk: each thread either does one exec-masked b128
// load (chunk fully in range) or zero-fills (fully out of range) -- no
// per-element loads even when N < BN. Software-pipelined vs the WMMAs.
// ---------------------------------------------------------------------------
#define BM 128
#define BN 64
#define BK 32
#define LDS_S 40   // padded ushort stride (row = 80B, keeps b128 runs 16B aligned)

__global__ __launch_bounds__(256) void k_gemm_bf16(
    const unsigned short* __restrict__ A,   // [M,K] bf16 row-major
    const unsigned short* __restrict__ B,   // [K,N] bf16 row-major
    float* __restrict__ C,                  // [M,N] fp32
    unsigned short* __restrict__ Cbf,       // optional bf16 mirror (may be null)
    int M, int N, int K,
    const float* __restrict__ bias,         // [N] or null
    const float* __restrict__ rowscale,     // [M] or null
    int leaky)
{
    __shared__ __align__(16) unsigned short As[BM * LDS_S];  // As[row][k]
    __shared__ __align__(16) unsigned short Bs[BN * LDS_S];  // Bs[col][k] (transposed)

    const int tid    = threadIdx.x;
    const int wave   = tid >> 5;
    const int lane   = tid & 31;
    const int laneHi = lane >> 4;
    const int lane16 = lane & 15;
    const int wr = wave >> 1;            // 0..3 : rows wr*32..+32
    const int wc = wave & 1;             // 0..1 : cols wc*32..+32
    const int rowBase = blockIdx.y * BM;
    const int colBase = blockIdx.x * BN;

    // per-thread tile-fill coordinates (A: 2 chunks of 8, B: 1 chunk of 8)
    const int arr = tid >> 2, ac0 = (tid & 3) * 8;   // A rows arr & arr+64
    const int bkk = tid >> 3, bc0 = (tid & 7) * 8;   // B: 32 k x 8 chunks
    const bool k8 = (K & 7) == 0;
    const bool n8 = (N & 7) == 0;
    const bool wactive = (rowBase + wr * 32 < M) && (colBase + wc * 32 < N);

    // Chunk-granular loads: one b128 if chunk fully in range, zeros if fully
    // out of range; scalar clamp+select only for ragged K/N tails (rare).
    auto loadArow = [&](int row, int k0) -> ushort8 {
        ushort8 v = {};
        int gr = rowBase + row;
        int gk = k0 + ac0;                 // multiple of 8
        if (k8) {
            if (gr < M && gk < K)
                v = *(const ushort8*)(A + (size_t)gr * K + gk);
        } else if (gr < M && gk < K) {
            unsigned short t[8];
            #pragma unroll
            for (int j = 0; j < 8; ++j) {
                int gkj = gk + j;
                int gkc = gkj < K ? gkj : K - 1;
                t[j] = A[(size_t)gr * K + gkc];
            }
            #pragma unroll
            for (int j = 0; j < 8; ++j)
                v[j] = (gk + j < K) ? t[j] : (unsigned short)0;
        }
        return v;
    };
    auto loadB = [&](int k0) -> ushort8 {
        ushort8 v = {};
        int gk = k0 + bkk;
        int gc = colBase + bc0;            // multiple of 8
        if (n8) {
            if (gk < K && gc < N)
                v = *(const ushort8*)(B + (size_t)gk * N + gc);
        } else if (gk < K && gc < N) {
            unsigned short t[8];
            #pragma unroll
            for (int j = 0; j < 8; ++j) {
                int gcj = gc + j;
                int gcc = gcj < N ? gcj : N - 1;
                t[j] = B[(size_t)gk * N + gcc];
            }
            #pragma unroll
            for (int j = 0; j < 8; ++j)
                v[j] = (gc + j < N) ? t[j] : (unsigned short)0;
        }
        return v;
    };

    v8f acc00 = {}, acc01 = {}, acc10 = {}, acc11 = {};
    ushort8 va0 = loadArow(arr,      0);
    ushort8 va1 = loadArow(arr + 64, 0);
    ushort8 vb  = loadB(0);

    for (int k0 = 0; k0 < K; k0 += BK) {
        // commit staged tile to LDS
        *(ushort8*)&As[arr * LDS_S + ac0]        = va0;
        *(ushort8*)&As[(arr + 64) * LDS_S + ac0] = va1;
        #pragma unroll
        for (int j = 0; j < 8; ++j) Bs[(bc0 + j) * LDS_S + bkk] = vb[j];
        __syncthreads();

        // issue next tile's global loads before consuming LDS (pipeline)
        int k1 = k0 + BK;
        if (k1 < K) {
            va0 = loadArow(arr,      k1);
            va1 = loadArow(arr + 64, k1);
            vb  = loadB(k1);
        }

        if (wactive) {
            // A fragments: rows wr*32 + rr*16 + lane16; 16-bit A 16x32 layout
            BF16x16 af0, af1, bf0, bf1;
            const int ar0 = wr * 32 + lane16;
            #pragma unroll
            for (int i = 0; i < 16; ++i) {
                int kk = i + (i >= 8 ? 8 : 0) + laneHi * 8;
                af0.u[i] = As[ar0 * LDS_S + kk];
                af1.u[i] = As[(ar0 + 16) * LDS_S + kk];
            }
            // B fragments: col = wc*32 + lane16 (+16), K = table order (same as A)
            const int bcA = wc * 32 + lane16;
            #pragma unroll
            for (int i = 0; i < 16; ++i) {
                int kk = i + (i >= 8 ? 8 : 0) + laneHi * 8;
                bf0.u[i] = Bs[bcA * LDS_S + kk];
                bf1.u[i] = Bs[(bcA + 16) * LDS_S + kk];
            }
            acc00 = __builtin_amdgcn_wmma_f32_16x16x32_bf16(false, af0.v, false, bf0.v,
                                                            (short)0, acc00, false, false);
            acc01 = __builtin_amdgcn_wmma_f32_16x16x32_bf16(false, af0.v, false, bf1.v,
                                                            (short)0, acc01, false, false);
            acc10 = __builtin_amdgcn_wmma_f32_16x16x32_bf16(false, af1.v, false, bf0.v,
                                                            (short)0, acc10, false, false);
            acc11 = __builtin_amdgcn_wmma_f32_16x16x32_bf16(false, af1.v, false, bf1.v,
                                                            (short)0, acc11, false, false);
        }
        __syncthreads();
    }

    if (!wactive) return;

    // Epilogue
    v8f accs[2][2] = { { acc00, acc01 }, { acc10, acc11 } };
    #pragma unroll
    for (int rr = 0; rr < 2; ++rr) {
        #pragma unroll
        for (int cc = 0; cc < 2; ++cc) {
            int col = colBase + wc * 32 + cc * 16 + lane16;
            if (col >= N) continue;
            #pragma unroll
            for (int j = 0; j < 8; ++j) {
                int row = rowBase + wr * 32 + rr * 16 + j + laneHi * 8;
                if (row >= M) continue;
                float v = accs[rr][cc][j];
                if (rowscale) v *= rowscale[row];
                if (bias)     v += bias[col];
                if (leaky)    v  = lrelu(v);
                C[(size_t)row * N + col] = v;
                if (Cbf) Cbf[(size_t)row * N + col] = f2bf(v);
            }
        }
    }
}

// ---------------------------------------------------------------------------
// KPConv aggregation: one workgroup (128 thr / 4 waves) per query point.
//   agg[n, kp, c] = sum_h relu(1 - |dp(h)-kpt(kp)|/sigma) * feats[nbr(h), c]
// W (15x40 -> 16x64 padded) and NF (40xC -> 64xCpad, stored [c][k]) in LDS,
// contracted with v_wmma_f32_16x16x32_bf16 (two K=32 steps). Neighbor gather
// is b128-vectorized (8 channels per load); the valid-neighbor count is
// computed from the LDS tile (no duplicate global traffic).
// ---------------------------------------------------------------------------
__global__ __launch_bounds__(128) void k_kpconv_agg(
    const unsigned short* __restrict__ feats,  // [Ns, C] bf16
    const float* __restrict__ qpts,            // [Nq, 3]
    const float* __restrict__ spts,            // [Ns, 3]
    const int*   __restrict__ nbr,             // [Nq, 40]
    const float* __restrict__ kpts,            // [15, 3]
    float inv_sigma, int C,
    unsigned short* __restrict__ agg,          // [Nq, 15*C] bf16
    float* __restrict__ rowscale)              // [Nq]
{
    __shared__ __align__(16) unsigned short NFs[256 * 64];   // [c][k]
    __shared__ __align__(16) unsigned short Ws[16 * 64];     // [m][k]
    __shared__ float dpx[64], dpy[64], dpz[64];
    __shared__ int   idxs[64];
    __shared__ int   cnt;

    const int n   = blockIdx.x;
    const int tid = threadIdx.x;
    if (tid == 0) cnt = 0;

    const float qx = qpts[n * 3 + 0];
    const float qy = qpts[n * 3 + 1];
    const float qz = qpts[n * 3 + 2];

    if (tid < 40) {
        int id = nbr[n * 40 + tid];
        idxs[tid] = id;
        dpx[tid] = spts[id * 3 + 0] - qx;
        dpy[tid] = spts[id * 3 + 1] - qy;
        dpz[tid] = spts[id * 3 + 2] - qz;
    } else if (tid < 64) {
        idxs[tid] = 0; dpx[tid] = 0.f; dpy[tid] = 0.f; dpz[tid] = 0.f;
    }
    __syncthreads();

    const int Cpad = (C + 15) & ~15;
    if ((C & 7) == 0) {
        // vector gather: chunk = (c0/8, k); 8 contiguous channels per b128 load
        const int nchunk = (C >> 3) * 64;
        for (int e = tid; e < nchunk; e += 128) {
            int k  = e & 63;
            int c0 = (e >> 6) * 8;
            ushort8 v = {};
            if (k < 40) v = *(const ushort8*)(feats + (size_t)idxs[k] * C + c0);
            #pragma unroll
            for (int j = 0; j < 8; ++j) NFs[(c0 + j) * 64 + k] = v[j];
        }
    } else {
        for (int e = tid; e < Cpad * 64; e += 128) {
            int c = e >> 6, k = e & 63;
            unsigned short v = 0;
            if (k < 40 && c < C) v = feats[(size_t)idxs[k] * C + c];
            NFs[c * 64 + k] = v;
        }
    }
    for (int e = tid; e < 16 * 64; e += 128) {
        int m = e >> 6, k = e & 63;
        unsigned short v = 0;
        if (m < 15 && k < 40) {
            float dx = dpx[k] - kpts[m * 3 + 0];
            float dy = dpy[k] - kpts[m * 3 + 1];
            float dz = dpz[k] - kpts[m * 3 + 2];
            float w = 1.f - sqrtf(dx * dx + dy * dy + dz * dz) * inv_sigma;
            v = f2bf(w > 0.f ? w : 0.f);
        }
        Ws[m * 64 + k] = v;
    }
    __syncthreads();

    // valid-neighbor count from the LDS tile
    if (tid < 40) {
        float s = 0.f;
        for (int c = 0; c < C; ++c) s += bf2f(NFs[c * 64 + tid]);
        if (s > 0.f) atomicAdd(&cnt, 1);
    }
    __syncthreads();
    if (tid == 0) rowscale[n] = 1.f / (float)(cnt < 1 ? 1 : cnt);

    const int wave = tid >> 5, lane = tid & 31;
    const int laneHi = lane >> 4, lane16 = lane & 15;

    BF16x16 a0, a1;
    #pragma unroll
    for (int i = 0; i < 16; ++i) {
        int kk = i + (i >= 8 ? 8 : 0) + laneHi * 8;
        a0.u[i] = Ws[lane16 * 64 + kk];
        a1.u[i] = Ws[lane16 * 64 + 32 + kk];
    }

    const int nt = Cpad >> 4;
    for (int ct = wave; ct < nt; ct += 4) {
        BF16x16 b0, b1;
        int col = ct * 16 + lane16;
        #pragma unroll
        for (int i = 0; i < 16; ++i) {
            int kk = i + (i >= 8 ? 8 : 0) + laneHi * 8;
            b0.u[i] = NFs[col * 64 + kk];
            b1.u[i] = NFs[col * 64 + 32 + kk];
        }
        v8f acc = {};
        acc = __builtin_amdgcn_wmma_f32_16x16x32_bf16(false, a0.v, false, b0.v,
                                                      (short)0, acc, false, false);
        acc = __builtin_amdgcn_wmma_f32_16x16x32_bf16(false, a1.v, false, b1.v,
                                                      (short)0, acc, false, false);
        #pragma unroll
        for (int j = 0; j < 8; ++j) {
            int m = j + laneHi * 8;
            int c = ct * 16 + lane16;
            if (m < 15 && c < C)
                agg[(size_t)n * 15 * C + (size_t)m * C + c] = f2bf(acc[j]);
        }
    }
}

// ---------------------------------------------------------------------------
// GroupNorm (32 groups) [+ optional leaky] -> fp32 out (+ optional bf16 out)
// One 32-thread wave per row; thread = one group.
// ---------------------------------------------------------------------------
__global__ void k_groupnorm(const float* __restrict__ x, float* __restrict__ out,
                            unsigned short* __restrict__ outbf,
                            const float* __restrict__ g, const float* __restrict__ be,
                            int N, int C, int leaky)
{
    int row = blockIdx.x * (blockDim.x >> 5) + (threadIdx.x >> 5);
    int grp = threadIdx.x & 31;
    if (row >= N) return;
    int cpg = C >> 5;
    const float* xr = x + (size_t)row * C + grp * cpg;
    float s = 0.f, s2 = 0.f;
    for (int i = 0; i < cpg; ++i) { float v = xr[i]; s += v; s2 += v * v; }
    float m   = s / (float)cpg;
    float var = s2 / (float)cpg - m * m;
    float inv = rsqrtf(var + 1e-5f);
    for (int i = 0; i < cpg; ++i) {
        int c = grp * cpg + i;
        float v = (xr[i] - m) * inv * g[c] + be[c];
        if (leaky) v = lrelu(v);
        out[(size_t)row * C + c] = v;
        if (outbf) outbf[(size_t)row * C + c] = f2bf(v);
    }
}

// out = leaky(a + b)  (+ optional bf16 mirror), vec4
__global__ void k_add_leaky4(const float* __restrict__ a, const float* __restrict__ b,
                             float* __restrict__ out, unsigned short* __restrict__ outbf, int n4)
{
    int i = blockIdx.x * blockDim.x + threadIdx.x;
    if (i < n4) {
        float4 va = *(const float4*)(a + (size_t)i * 4);
        float4 vb = *(const float4*)(b + (size_t)i * 4);
        float4 v;
        v.x = lrelu(va.x + vb.x); v.y = lrelu(va.y + vb.y);
        v.z = lrelu(va.z + vb.z); v.w = lrelu(va.w + vb.w);
        *(float4*)(out + (size_t)i * 4) = v;
        if (outbf) {
            u16x4 o;
            o[0] = f2bf(v.x); o[1] = f2bf(v.y); o[2] = f2bf(v.z); o[3] = f2bf(v.w);
            *(u16x4*)(outbf + (size_t)i * 4) = o;
        }
    }
}

// strided shortcut: out[n, 4c..4c+3] = max_h x[nbr[n,h], 4c..4c+3]  (vec4)
__global__ void k_maxpool4(const float* __restrict__ x, const int* __restrict__ nbr,
                           float* __restrict__ out, int N, int C, int Hn)
{
    int i = blockIdx.x * blockDim.x + threadIdx.x;
    int C4 = C >> 2;
    if (i < N * C4) {
        int r = i / C4, c4 = (i % C4) * 4;
        float4 m = make_float4(-1e30f, -1e30f, -1e30f, -1e30f);
        for (int h = 0; h < Hn; ++h) {
            int id = nbr[r * Hn + h];
            float4 v = *(const float4*)(x + (size_t)id * C + c4);
            m.x = v.x > m.x ? v.x : m.x;
            m.y = v.y > m.y ? v.y : m.y;
            m.z = v.z > m.z ? v.z : m.z;
            m.w = v.w > m.w ? v.w : m.w;
        }
        *(float4*)(out + (size_t)r * C + c4) = m;
    }
}

// dst[r, dstOff + c (stride dstStride)] = src[(idx?idx[r]:r), c]   (vec4)
__global__ void k_gather_rows4(const float* __restrict__ src, const int* __restrict__ idx,
                               float* __restrict__ dst, int N, int C, int dstStride, int dstOff)
{
    int i = blockIdx.x * blockDim.x + threadIdx.x;
    int C4 = C >> 2;
    if (i < N * C4) {
        int r = i / C4, c4 = (i % C4) * 4;
        int sr = idx ? idx[r] : r;
        *(float4*)(dst + (size_t)r * dstStride + dstOff + c4) =
            *(const float4*)(src + (size_t)sr * C + c4);
    }
}

// ---------------------------------------------------------------------------
// Host orchestration
// ---------------------------------------------------------------------------
extern "C" void kernel_launch(void* const* d_in, const int* in_sizes, int n_in,
                              void* d_out, int out_size, void* d_ws, size_t ws_size,
                              hipStream_t stream) {
    (void)in_sizes; (void)n_in; (void)out_size; (void)ws_size;
    const int N0 = 40000, N1 = 10000, N2 = 2500, N3 = 625, H = 40;

    auto F = [&](int i) { return (const float*)d_in[i]; };
    auto I = [&](int i) { return (const int*)d_in[i]; };

    int pi = 0;
    const float* feats = F(pi++);
    const float* P[4]; const int* NB[4];
    for (int i = 0; i < 4; ++i) { P[i] = F(pi++); NB[i] = I(pi++); }
    const int* SB[3]; for (int i = 0; i < 3; ++i) SB[i] = I(pi++);
    const int* UP1 = I(pi++); const int* UP2 = I(pi++);

    const float* e11_kpw = F(pi++); const float* e11_g = F(pi++); const float* e11_be = F(pi++);

    struct ResP {
        const float *u1W,*u1b,*u1g,*u1be,*kpw,*g,*be,*u2W,*u2b,*u2g,*u2be,*scW,*scb,*scg,*scbe;
        int cin, cout, h;
        const unsigned short *u1Wb,*kpwb,*u2Wb,*scWb;
    };
    auto loadRes = [&](int cin, int cout, bool sc) {
        ResP r{}; r.cin = cin; r.cout = cout; r.h = cout / 4;
        r.u1W = F(pi++); r.u1b = F(pi++); r.u1g = F(pi++); r.u1be = F(pi++);
        r.kpw = F(pi++); r.g = F(pi++); r.be = F(pi++);
        r.u2W = F(pi++); r.u2b = F(pi++); r.u2g = F(pi++); r.u2be = F(pi++);
        if (sc) { r.scW = F(pi++); r.scb = F(pi++); r.scg = F(pi++); r.scbe = F(pi++); }
        return r;
    };
    ResP e12 = loadRes( 64, 128, true ),  e21 = loadRes(128, 128, false);
    ResP e22 = loadRes(128, 256, true ),  e23 = loadRes(256, 256, false);
    ResP e31 = loadRes(256, 256, false),  e32 = loadRes(256, 512, true );
    ResP e33 = loadRes(512, 512, false),  e41 = loadRes(512, 512, false);
    ResP e42 = loadRes(512,1024, true ),  e43 = loadRes(1024,1024,false);
    const float* dec3W = F(pi++); const float* dec3b = F(pi++);
    const float* dec3g = F(pi++); const float* dec3be = F(pi++);
    const float* dec2W = F(pi++); const float* dec2b = F(pi++);
    const float* KP[11]; for (int i = 0; i < 11; ++i) KP[i] = F(pi++);

    // ----- workspace arena -----
    char* base = (char*)d_ws; size_t off = 0;
    auto aalloc = [&](size_t bytes) -> void* {
        void* p = base + off; off += (bytes + 255) & ~(size_t)255; return p;
    };
    float* T0 = (float*)aalloc((size_t)7680000 * 4);           // [N1,768] max / [N0,128]
    float* T1 = (float*)aalloc((size_t)7680000 * 4);
    float* T2 = (float*)aalloc((size_t)5120000 * 4);           // [N0,128] max
    unsigned short* XA  = (unsigned short*)aalloc((size_t)8000000 * 2);
    unsigned short* XB  = (unsigned short*)aalloc((size_t)2600000 * 2);
    unsigned short* AGG = (unsigned short*)aalloc((size_t)19200000 * 2); // [N0,15*32]
    float* RS = (float*)aalloc((size_t)N0 * 4);
    unsigned short* WAR = (unsigned short*)aalloc((size_t)6400000 * 2);
    unsigned short* wcur = WAR;

    // ----- d_out slices -----
    float* out = (float*)d_out;
    float* L2 = out;                 // [N1,256]
    float* L3 = out + 2560000;       // [N2,512]
    float* F4 = out + 3840000;       // [N3,1024]
    float* F1 = out + 4480000;       // [N0,128]
    float* F2 = out + 9600000;       // [N1,256]
    float* F3 = out + 12160000;      // [N2,512]

    // ----- launch helpers -----
    auto cvt = [&](const float* s, unsigned short* d, size_t n) {
        size_t n4 = n >> 2;   // all sizes here are multiples of 4
        k_cvt4<<<dim3((unsigned)((n4 + 255) / 256)), 256, 0, stream>>>(s, d, (int)n4);
    };
    auto cvtw = [&](const float* s, size_t n) -> const unsigned short* {
        unsigned short* d = wcur; wcur += (n + 15) & ~(size_t)15;
        cvt(s, d, n); return d;
    };
    auto gemm = [&](const unsigned short* A, const unsigned short* B, float* C,
                    unsigned short* Cbf, int M, int N, int K,
                    const float* bias, const float* rs, int leaky) {
        dim3 g((N + BN - 1) / BN, (M + BM - 1) / BM);
        k_gemm_bf16<<<g, 256, 0, stream>>>(A, B, C, Cbf, M, N, K, bias, rs, leaky);
    };
    auto gnorm = [&](const float* x, float* o, unsigned short* obf,
                     const float* g, const float* be, int N, int C, int leaky) {
        k_groupnorm<<<dim3((N + 7) / 8), 256, 0, stream>>>(x, o, obf, g, be, N, C, leaky);
    };
    auto kpagg = [&](const unsigned short* fbf, const float* q, const float* s,
                     const int* nb, const float* kpt, float sigma, int Nq, int C) {
        k_kpconv_agg<<<dim3(Nq), 128, 0, stream>>>(fbf, q, s, nb, kpt, 1.f / sigma, C, AGG, RS);
    };

    // convert all GEMM weights to bf16 (every call; deterministic)
    const unsigned short* kw11 = cvtw(e11_kpw, 15 * 1 * 64);
    ResP* blocks[10] = { &e12,&e21,&e22,&e23,&e31,&e32,&e33,&e41,&e42,&e43 };
    for (int i = 0; i < 10; ++i) {
        ResP& r = *blocks[i];
        r.u1Wb = cvtw(r.u1W, (size_t)r.cin * r.h);
        r.kpwb = cvtw(r.kpw, (size_t)15 * r.h * r.h);
        r.u2Wb = cvtw(r.u2W, (size_t)r.h * r.cout);
        r.scWb = r.scW ? cvtw(r.scW, (size_t)r.cin * r.cout) : nullptr;
    }
    const unsigned short* dec3Wb = cvtw(dec3W, (size_t)1536 * 512);
    const unsigned short* dec2Wb = cvtw(dec2W, (size_t)768 * 256);

    // ----- residual block driver -----
    auto residual = [&](ResP& r, const float* xf, const unsigned short* xbf,
                        const float* q, const float* s, const int* nb, const float* kpt,
                        float sigma, int Nin, int Nq, bool strided,
                        float* of, unsigned short* obf) {
        // u1: Cin -> h, GN + leaky
        gemm(xbf, r.u1Wb, T0, nullptr, Nin, r.h, r.cin, r.u1b, nullptr, 0);
        gnorm(T0, T0, XB, r.u1g, r.u1be, Nin, r.h, 1);
        // KPConv h -> h  (agg via WMMA, then GEMM with 1/n rowscale), GN + leaky
        kpagg(XB, q, s, nb, kpt, sigma, Nq, r.h);
        gemm(AGG, r.kpwb, T0, nullptr, Nq, r.h, 15 * r.h, nullptr, RS, 0);
        gnorm(T0, T0, XB, r.g, r.be, Nq, r.h, 1);
        // u2: h -> Cout, GN (no act)
        gemm(XB, r.u2Wb, T0, nullptr, Nq, r.cout, r.h, r.u2b, nullptr, 0);
        gnorm(T0, T0, nullptr, r.u2g, r.u2be, Nq, r.cout, 0);
        // shortcut
        const float* sc = xf;
        if (strided) {
            int n4 = (Nq * r.cin) >> 2;
            k_maxpool4<<<dim3((n4 + 255) / 256), 256, 0, stream>>>(xf, nb, T2, Nq, r.cin, H);
            sc = T2;
        }
        if (r.scW) {
            gemm(xbf, r.scWb, T2, nullptr, Nq, r.cout, r.cin, r.scb, nullptr, 0);
            gnorm(T2, T2, nullptr, r.scg, r.scbe, Nq, r.cout, 0);
            sc = T2;
        }
        int n4 = (Nq * r.cout) >> 2;
        k_add_leaky4<<<dim3((n4 + 255) / 256), 256, 0, stream>>>(T0, sc, of, obf, n4);
    };

    const float sg = 0.05f;

    // ----- e11: KPConv(1 -> 64) + GN + leaky -----
    cvt(feats, XA, (size_t)N0);
    kpagg(XA, P[0], P[0], NB[0], KP[0], sg, N0, 1);
    gemm(AGG, kw11, T0, nullptr, N0, 64, 15, nullptr, RS, 0);
    gnorm(T0, T1, XA, e11_g, e11_be, N0, 64, 1);          // f1-pre: fp32 T1, bf16 XA

    // ----- encoder -----
    residual(e12, T1, XA, P[0], P[0], NB[0], KP[1],     sg, N0, N0, false, F1, XA); // f1
    residual(e21, F1, XA, P[1], P[0], SB[0], KP[2],     sg, N0, N1, true,  T1, XA);
    residual(e22, T1, XA, P[1], P[1], NB[1], KP[3], 2 * sg, N1, N1, false, T1, XA);
    residual(e23, T1, XA, P[1], P[1], NB[1], KP[4], 2 * sg, N1, N1, false, F2, XA); // f2
    residual(e31, F2, XA, P[2], P[1], SB[1], KP[5], 2 * sg, N1, N2, true,  T1, XA);
    residual(e32, T1, XA, P[2], P[2], NB[2], KP[6], 4 * sg, N2, N2, false, T1, XA);
    residual(e33, T1, XA, P[2], P[2], NB[2], KP[7], 4 * sg, N2, N2, false, F3, XA); // f3
    residual(e41, F3, XA, P[3], P[2], SB[2], KP[8], 4 * sg, N2, N3, true,  T1, XA);
    residual(e42, T1, XA, P[3], P[3], NB[3], KP[9], 8 * sg, N3, N3, false, T1, XA);
    residual(e43, T1, XA, P[3], P[3], NB[3], KP[10],8 * sg, N3, N3, false, F4, XA); // f4

    // ----- dec3: unary(concat(up(f4), f3) [N2,1536] -> 512) -----
    {
        int n4 = (N2 * 1024) >> 2;
        k_gather_rows4<<<dim3((n4 + 255) / 256), 256, 0, stream>>>(F4, UP2, T0, N2, 1024, 1536, 0);
        n4 = (N2 * 512) >> 2;
        k_gather_rows4<<<dim3((n4 + 255) / 256), 256, 0, stream>>>(F3, nullptr, T0, N2, 512, 1536, 1024);
        cvt(T0, XA, (size_t)N2 * 1536);
        gemm(XA, dec3Wb, T1, nullptr, N2, 512, 1536, dec3b, nullptr, 0);
        gnorm(T1, L3, nullptr, dec3g, dec3be, N2, 512, 1);     // l3
    }

    // ----- dec2: concat(up(l3), f2) [N1,768] @ dec2W + b -----
    {
        int n4 = (N1 * 512) >> 2;
        k_gather_rows4<<<dim3((n4 + 255) / 256), 256, 0, stream>>>(L3, UP1, T0, N1, 512, 768, 0);
        n4 = (N1 * 256) >> 2;
        k_gather_rows4<<<dim3((n4 + 255) / 256), 256, 0, stream>>>(F2, nullptr, T0, N1, 256, 768, 512);
        cvt(T0, XA, (size_t)N1 * 768);
        gemm(XA, dec2Wb, L2, nullptr, N1, 256, 768, dec2b, nullptr, 0);   // l2
    }
}
